// HGACLoss_29583734735620
// MI455X (gfx1250) — matrix operand order repaired and checked by reading, not compiled, not run on hardware
//
#include <hip/hip_runtime.h>
#include <hip/hip_bf16.h>
#include <math.h>

#define TEMP_INV (1.0f / 0.07f)
#define BB 128
#define NN 128
#define DD 1024

typedef __attribute__((ext_vector_type(16))) __bf16 v16bf;
typedef __attribute__((ext_vector_type(8)))  float  v8f;
typedef __attribute__((ext_vector_type(4)))  float  f32x4;
typedef __attribute__((ext_vector_type(2)))  float  v2f;

union FragBF {
    f32x4 f4[2];
    v16bf v;
};

// ---------------- accumulator init ----------------
__global__ void init_accum_kernel(float* accum) {
    if (threadIdx.x < 8) accum[threadIdx.x] = 0.0f;
}

// ---------------- row L2 normalize -> f32 ----------------
__global__ void normalize_f32_kernel(const float* __restrict__ X, float* __restrict__ Y, int D) {
    __shared__ float red[256];
    size_t row = blockIdx.x;
    const float* x = X + row * (size_t)D;
    float* y = Y + row * (size_t)D;
    float s = 0.0f;
    for (int d = threadIdx.x; d < D; d += blockDim.x) { float v = x[d]; s = fmaf(v, v, s); }
    red[threadIdx.x] = s;
    __syncthreads();
    for (int t = blockDim.x >> 1; t > 0; t >>= 1) {
        if ((int)threadIdx.x < t) red[threadIdx.x] += red[threadIdx.x + t];
        __syncthreads();
    }
    float scale = 1.0f / fmaxf(sqrtf(red[0]), 1e-12f);
    for (int d = threadIdx.x; d < D; d += blockDim.x) y[d] = x[d] * scale;
}

// ---------------- row L2 normalize -> bf16 ----------------
__global__ void normalize_bf16_kernel(const float* __restrict__ X, __bf16* __restrict__ Y, int D) {
    __shared__ float red[256];
    size_t row = blockIdx.x;
    const float* x = X + row * (size_t)D;
    __bf16* y = Y + row * (size_t)D;
    float s = 0.0f;
    for (int d = threadIdx.x; d < D; d += blockDim.x) { float v = x[d]; s = fmaf(v, v, s); }
    red[threadIdx.x] = s;
    __syncthreads();
    for (int t = blockDim.x >> 1; t > 0; t >>= 1) {
        if ((int)threadIdx.x < t) red[threadIdx.x] += red[threadIdx.x + t];
        __syncthreads();
    }
    float scale = 1.0f / fmaxf(sqrtf(red[0]), 1e-12f);
    for (int d = threadIdx.x; d < D; d += blockDim.x) y[d] = (__bf16)(x[d] * scale);
}

// ---------------- global sim matrix via f32 WMMA (16x16x4) ----------------
// sim = (vn @ tn^T) / TEMP, full f32 precision. One wave per 16x16 tile.
__global__ void __launch_bounds__(32) sim_wmma_kernel(const float* __restrict__ vn,
                                                      const float* __restrict__ tn,
                                                      float* __restrict__ sim) {
    const int B = BB, D = DD;
    int ti = blockIdx.x >> 3;          // row tile
    int tj = blockIdx.x & 7;           // col tile
    int lane = threadIdx.x;
    int l15 = lane & 15;
    int lh  = lane >> 4;               // 0 or 1
    // A (16x4 f32): lane = M, VGPR r: K = 2*lh + r
    const float* pa = vn + (size_t)(ti * 16 + l15) * D + 2 * lh;
    // B (4x16 f32): lane = N, VGPR r: K = 2*lh + r ; B[k][n] = tn[n][k]
    const float* pb = tn + (size_t)(tj * 16 + l15) * D + 2 * lh;
    v8f acc = {0.f, 0.f, 0.f, 0.f, 0.f, 0.f, 0.f, 0.f};
    for (int k = 0; k < D; k += 4) {
        v2f a = *reinterpret_cast<const v2f*>(pa + k);
        v2f b = *reinterpret_cast<const v2f*>(pb + k);
        acc = __builtin_amdgcn_wmma_f32_16x16x4_f32(false, a, false, b,
                                                    (short)0, acc, false, false);
    }
    int n = tj * 16 + l15;
#pragma unroll
    for (int r = 0; r < 8; ++r) {
        int m = ti * 16 + r + (lh << 3);
        sim[(size_t)m * B + n] = acc[r] * TEMP_INV;
    }
}

// ---------------- InfoNCE on sim: one block of B threads ----------------
__global__ void global_loss_kernel(const float* __restrict__ sim, float* __restrict__ accum, int B) {
    __shared__ float red[BB];
    int i = threadIdx.x;
    float mr = -__builtin_inff(), mc = -__builtin_inff();
    for (int j = 0; j < B; ++j) {
        mr = fmaxf(mr, sim[(size_t)i * B + j]);
        mc = fmaxf(mc, sim[(size_t)j * B + i]);
    }
    float sr = 0.0f, sc = 0.0f;
    for (int j = 0; j < B; ++j) {
        sr += expf(sim[(size_t)i * B + j] - mr);
        sc += expf(sim[(size_t)j * B + i] - mc);
    }
    float lse_r = mr + logf(sr);
    float lse_c = mc + logf(sc);
    float diag = sim[(size_t)i * B + i];
    red[i] = 0.5f * ((lse_r - diag) + (lse_c - diag));
    __syncthreads();
    for (int t = B >> 1; t > 0; t >>= 1) {
        if (i < t) red[i] += red[i + t];
        __syncthreads();
    }
    if (i == 0) accum[2] = red[0] / (float)B;
}

// ---------------- batched cost GEMM via WMMA bf16 ----------------
// cost[b] = 1 - Vn[b] (NxD) * Tn[b]^T (DxN).  One block (8 waves) per batch.
__global__ void __launch_bounds__(256) cost_wmma_kernel(const __bf16* __restrict__ Vn,
                                                        const __bf16* __restrict__ Tn,
                                                        float* __restrict__ cost) {
    const int N = NN, D = DD;
    int b = blockIdx.x;
    int wave = threadIdx.x >> 5;
    int lane = threadIdx.x & 31;
    int lhalf = lane >> 4;
    int l15 = lane & 15;
    const __bf16* Vb = Vn + (size_t)b * N * D;
    const __bf16* Tb = Tn + (size_t)b * N * D;
    float* Cb = cost + (size_t)b * N * N;

    int mrow = wave * 16 + l15;
    int ka = lhalf << 3;
    for (int t = 0; t < N; t += 16) {
        int nrow = t + l15;
        int kb = lhalf << 3;
        v8f acc = {0.f, 0.f, 0.f, 0.f, 0.f, 0.f, 0.f, 0.f};
        for (int k = 0; k < D; k += 32) {
            FragBF fa, fb;
            const __bf16* pa = Vb + (size_t)mrow * D + k + ka;
            fa.f4[0] = *reinterpret_cast<const f32x4*>(pa);
            fa.f4[1] = *reinterpret_cast<const f32x4*>(pa + 16);
            const __bf16* pb = Tb + (size_t)nrow * D + k + kb;
            fb.f4[0] = *reinterpret_cast<const f32x4*>(pb);
            fb.f4[1] = *reinterpret_cast<const f32x4*>(pb + 16);
            acc = __builtin_amdgcn_wmma_f32_16x16x32_bf16(
                false, fa.v, false, fb.v, (short)0, acc, false, false);
        }
        int ncol = t + l15;
        int mbase = wave * 16 + (lhalf << 3);
#pragma unroll
        for (int r = 0; r < 8; ++r)
            Cb[(size_t)(mbase + r) * N + ncol] = 1.0f - acc[r];
    }
}

// ---------------- JV Hungarian: ONE WAVE (32 lanes) per batch ----------------
// Each lane owns 4 columns: jj = 1 + lane + 32*q. minv / v / used in registers,
// argmin via shfl_xor butterfly, cost matrix staged in (dynamic) LDS.
__global__ void __launch_bounds__(32) hungarian_kernel(const float* __restrict__ cost,
                                                       int* __restrict__ cols) {
    const int n = NN;
    extern __shared__ char smem[];
    float* c   = (float*)smem;                                   // n*n f32
    float* u   = (float*)(smem + (size_t)n * n * 4);             // n+1
    int*   p   = (int*)  (smem + (size_t)n * n * 4 + (n + 1) * 4);
    int*   way = (int*)  (smem + (size_t)n * n * 4 + 2 * (n + 1) * 4);

    int b = blockIdx.x;
    int lane = threadIdx.x;
    const f32x4* Cg4 = (const f32x4*)(cost + (size_t)b * n * n);
    f32x4* c4 = (f32x4*)c;
    for (int idx = lane; idx < (n * n) / 4; idx += 32) c4[idx] = Cg4[idx];
    for (int idx = lane; idx < n + 1; idx += 32) { u[idx] = 0.0f; p[idx] = 0; way[idx] = 0; }
    __syncthreads();

    float vreg[4];                          // v[1 + lane + 32q]
#pragma unroll
    for (int q = 0; q < 4; ++q) vreg[q] = 0.0f;

    for (int i = 1; i <= n; ++i) {
        if (lane == 0) p[0] = i;
        float minv[4];
        int usedq = 0;
#pragma unroll
        for (int q = 0; q < 4; ++q) minv[q] = __builtin_inff();
        int j0 = 0;
        __syncthreads();
        while (true) {
            // mark used[j0] (column 0 is implicitly always "used")
            if (j0 > 0 && lane == ((j0 - 1) & 31)) usedq |= 1 << ((j0 - 1) >> 5);
            int i0 = p[j0];
            float ui0 = u[i0];
            const float* crow = c + (size_t)(i0 - 1) * n;
            float bestv = __builtin_inff();
            int bestj = 0;
#pragma unroll
            for (int q = 0; q < 4; ++q) {
                int col = lane + 32 * q;                 // 0-based column
                if (!(usedq & (1 << q))) {
                    float cur = crow[col] - ui0 - vreg[q];
                    if (cur < minv[q]) { minv[q] = cur; way[col + 1] = j0; }
                    if (minv[q] < bestv) { bestv = minv[q]; bestj = col + 1; }
                }
            }
            // wave32 butterfly min-reduce (value, index) -> uniform in all lanes
#pragma unroll
            for (int s = 16; s > 0; s >>= 1) {
                float ov = __shfl_xor(bestv, s, 32);
                int   oj = __shfl_xor(bestj, s, 32);
                if (ov < bestv) { bestv = ov; bestj = oj; }
            }
            float delta = bestv;
            int j1 = bestj;
            // potential updates (distinct u[] addresses across lanes)
#pragma unroll
            for (int q = 0; q < 4; ++q) {
                if (usedq & (1 << q)) {
                    int pj = p[lane + 32 * q + 1];
                    u[pj] += delta;
                    vreg[q] -= delta;
                } else {
                    minv[q] -= delta;
                }
            }
            if (lane == 0) u[p[0]] += delta;             // column 0 ("used")
            __syncthreads();
            j0 = j1;
            if (p[j0] == 0) break;
        }
        if (lane == 0) {                                 // augment alternating path
            int jj = j0;
            while (jj != 0) { int j1 = way[jj]; p[jj] = p[j1]; jj = j1; }
        }
        __syncthreads();
    }
#pragma unroll
    for (int q = 0; q < 4; ++q) {
        int col = lane + 32 * q;
        cols[(size_t)b * n + (p[col + 1] - 1)] = col;
    }
}

// ---------------- node MSE with gathered columns ----------------
__global__ void node_mse_kernel(const float* __restrict__ V, const float* __restrict__ T,
                                const int* __restrict__ cols, float* __restrict__ accum,
                                int N, int D) {
    __shared__ float red[256];
    size_t row = blockIdx.x;
    int b = (int)(row / N);
    int nn = (int)(row % N);
    int c = cols[(size_t)b * N + nn];
    const float* pv = V + ((size_t)b * N + nn) * D;
    const float* pt = T + ((size_t)b * N + c) * D;
    float s = 0.0f;
    for (int d = threadIdx.x; d < D; d += blockDim.x) {
        float diff = pv[d] - pt[d];
        s = fmaf(diff, diff, s);
    }
    red[threadIdx.x] = s;
    __syncthreads();
    for (int t = blockDim.x >> 1; t > 0; t >>= 1) {
        if ((int)threadIdx.x < t) red[threadIdx.x] += red[threadIdx.x + t];
        __syncthreads();
    }
    if (threadIdx.x == 0) atomicAdd(accum + 0, red[0]);
}

// ---------------- graph MSE ----------------
__global__ void graph_mse_kernel(const float* __restrict__ Av, const float* __restrict__ At,
                                 float* __restrict__ accum, size_t total) {
    __shared__ float red[256];
    float s = 0.0f;
    for (size_t idx = (size_t)blockIdx.x * blockDim.x + threadIdx.x; idx < total;
         idx += (size_t)gridDim.x * blockDim.x) {
        float diff = Av[idx] - At[idx];
        s = fmaf(diff, diff, s);
    }
    red[threadIdx.x] = s;
    __syncthreads();
    for (int t = blockDim.x >> 1; t > 0; t >>= 1) {
        if ((int)threadIdx.x < t) red[threadIdx.x] += red[threadIdx.x + t];
        __syncthreads();
    }
    if (threadIdx.x == 0) atomicAdd(accum + 1, red[0]);
}

// ---------------- finalize ----------------
__global__ void finalize_kernel(const float* __restrict__ accum, float* __restrict__ out) {
    float Lg  = accum[2];
    float Ln  = accum[0] / (float)((size_t)BB * NN * DD);
    float Lgr = accum[1] / (float)((size_t)BB * NN * NN);
    out[0] = Lg + Ln + Lgr;
    out[1] = Lg;
    out[2] = Ln;
    out[3] = Lgr;
}

extern "C" void kernel_launch(void* const* d_in, const int* in_sizes, int n_in,
                              void* d_out, int out_size, void* d_ws, size_t ws_size,
                              hipStream_t stream) {
    (void)in_sizes; (void)n_in; (void)out_size; (void)ws_size;
    const float* v_g = (const float*)d_in[0];
    const float* t_g = (const float*)d_in[1];
    const float* V   = (const float*)d_in[2];
    const float* T   = (const float*)d_in[3];
    const float* A_v = (const float*)d_in[4];
    const float* A_t = (const float*)d_in[5];
    float* out = (float*)d_out;

    char* ws = (char*)d_ws;
    size_t off = 0;
    __bf16* Vn  = (__bf16*)(ws + off); off += (size_t)BB * NN * DD * 2;   // 32 MB
    __bf16* Tn  = (__bf16*)(ws + off); off += (size_t)BB * NN * DD * 2;   // 32 MB
    float* cost = (float*) (ws + off); off += (size_t)BB * NN * NN * 4;   // 8 MB
    float* vng  = (float*) (ws + off); off += (size_t)BB * DD * 4;        // 512 KB
    float* tng  = (float*) (ws + off); off += (size_t)BB * DD * 4;        // 512 KB
    float* sim  = (float*) (ws + off); off += (size_t)BB * BB * 4;        // 64 KB
    int*   cols = (int*)   (ws + off); off += (size_t)BB * NN * 4;        // 64 KB
    float* accum= (float*) (ws + off); off += 64;

    init_accum_kernel<<<1, 32, 0, stream>>>(accum);

    normalize_f32_kernel<<<BB, 256, 0, stream>>>(v_g, vng, DD);
    normalize_f32_kernel<<<BB, 256, 0, stream>>>(t_g, tng, DD);
    normalize_bf16_kernel<<<BB * NN, 256, 0, stream>>>(V, Vn, DD);
    normalize_bf16_kernel<<<BB * NN, 256, 0, stream>>>(T, Tn, DD);

    sim_wmma_kernel<<<64, 32, 0, stream>>>(vng, tng, sim);
    global_loss_kernel<<<1, BB, 0, stream>>>(sim, accum, BB);

    cost_wmma_kernel<<<BB, 256, 0, stream>>>(Vn, Tn, cost);

    const size_t hung_smem = (size_t)NN * NN * 4 + 2 * (size_t)(NN + 1) * 4 + (size_t)(NN + 1) * 4;
    hungarian_kernel<<<BB, 32, hung_smem, stream>>>(cost, cols);

    node_mse_kernel<<<BB * NN, 256, 0, stream>>>(V, T, cols, accum, NN, DD);
    graph_mse_kernel<<<2048, 256, 0, stream>>>(A_v, A_t, accum, (size_t)BB * NN * NN);

    finalize_kernel<<<1, 1, 0, stream>>>(accum, out);
}
